// PointMLPEncoder_35098472743117
// MI455X (gfx1250) — matrix operand order repaired and checked by reading, not compiled, not run on hardware
//
#include <hip/hip_runtime.h>
#include <hip/hip_bf16.h>

// ---------------- types ----------------
typedef _Float16 h16 __attribute__((ext_vector_type(16)));
typedef _Float16 h8  __attribute__((ext_vector_type(8)));
typedef _Float16 h4  __attribute__((ext_vector_type(4)));
typedef float    f8  __attribute__((ext_vector_type(8)));
typedef float    f4  __attribute__((ext_vector_type(4)));

#define DEVFN static __device__ __forceinline__

// ---------------- problem constants ----------------
#define BATCH   4
#define NPTS    4096
#define PTOT    (BATCH*NPTS)   // 16384
#define DIM     128
#define DEPTH   6
#define KNN     16
#define HIDDEN  256
#define LN_EPS  1e-5f

// ---------------- WMMA helpers ----------------
DEVFN h16 cat8(h8 lo, h8 hi) {
    return __builtin_shufflevector(lo, hi, 0,1,2,3,4,5,6,7,8,9,10,11,12,13,14,15);
}

// A-fragment (16xK tile, 16-bit): lane m = lane&15, hi = lane>>4.
// Halves 0..7  -> K = k0 + hi*8 + 0..7
// Halves 8..15 -> K = k0 + 16 + hi*8 + 0..7      (ISA 7.12.2, 16-bit A 16x32)
DEVFN h16 load_a_frag(const _Float16* __restrict__ base, int ld, int lane, int k0) {
    int m  = lane & 15;
    int hi = lane >> 4;
    const _Float16* p = base + m * ld + k0 + hi * 8;
    h8 lo = *(const h8*)(p);
    h8 hh = *(const h8*)(p + 16);
    return cat8(lo, hh);
}

// A-fragment loaded directly from a per-lane row pointer (row already resolved)
DEVFN h16 load_a_frag_row(const _Float16* __restrict__ rowp, int lane, int k0) {
    int hi = lane >> 4;
    const _Float16* p = rowp + k0 + hi * 8;
    h8 lo = *(const h8*)(p);
    h8 hh = *(const h8*)(p + 16);
    return cat8(lo, hh);
}

// B-fragment (32x16 KxN tile): lane n = lane&15, hi = lane>>4; halves 0..7 -> K = k0 + hi*8 + 0..7, halves 8..15 -> K = k0 + 16 + hi*8 + 0..7
DEVFN h16 load_b_frag(const _Float16* __restrict__ W, int ldb, int lane, int k0, int n0) {
    int n  = lane & 15;
    int hi = lane >> 4;
    const _Float16* p = W + (size_t)(k0 + hi * 8) * ldb + n0 + n;
    h16 b;
#pragma unroll
    for (int s = 0; s < 8; ++s) {
        b[s]     = p[(size_t)s * ldb];
        b[8 + s] = p[(size_t)(16 + s) * ldb];
    }
    return b;
}

DEVFN f8 wmma16(h16 a, h16 b, f8 c) {
    return __builtin_amdgcn_wmma_f32_16x16x32_f16(false, a, false, b, (short)0, c, false, false);
}

// ---------------- weight prep ----------------
// msg_W1: [387][128] per layer, rows: 0..127 = W1a (h_i), 128..255 = W1b (neigh),
// 256..383 = W1c (neigh - h_i), 384..386 = W1d (rel_xyz).
// Wh = W1a - W1c  (128x128)   Wn = [W1b+W1c ; W1d ; 0-pad] (160x128)
__global__ void k_prep_msg1(const float* __restrict__ W,
                            _Float16* __restrict__ Wh, _Float16* __restrict__ Wn) {
    int i = blockIdx.x * 256 + threadIdx.x;
    if (i >= DEPTH * 160 * 128) return;
    int c = i & 127;
    int r = (i >> 7) % 160;
    int l = i / (160 * 128);
    const float* Wl = W + (size_t)l * 387 * 128;
    float wn;
    if (r < 128) {
        wn = Wl[(128 + r) * 128 + c] + Wl[(256 + r) * 128 + c];
        Wh[(size_t)l * 128 * 128 + r * 128 + c] =
            (_Float16)(Wl[r * 128 + c] - Wl[(256 + r) * 128 + c]);
    } else if (r < 131) {
        wn = Wl[(384 + (r - 128)) * 128 + c];
    } else {
        wn = 0.f;
    }
    Wn[(size_t)l * 160 * 128 + r * 128 + c] = (_Float16)wn;
}

__global__ void k_cast16(const float* __restrict__ s, _Float16* __restrict__ d, int n) {
    int i = blockIdx.x * 256 + threadIdx.x;
    if (i < n) d[i] = (_Float16)s[i];
}

// ---------------- KNN ----------------
// One block = 256 points of one batch; whole batch xyz staged in LDS (48 KB).
__global__ __launch_bounds__(256)
void k_knn(const float* __restrict__ pts, int* __restrict__ idx, _Float16* __restrict__ rel) {
    __shared__ float sx[NPTS], sy[NPTS], sz[NPTS];
    int tid  = threadIdx.x;
    int base = blockIdx.x * 256;
    int b    = base >> 12;
    int bb   = b << 12;
    for (int t = tid; t < NPTS; t += 256) {
        const float* pp = pts + (size_t)(bb + t) * 6;
        sx[t] = pp[0]; sy[t] = pp[1]; sz[t] = pp[2];
    }
    __syncthreads();
    int i  = base + tid;
    int il = i - bb;
    float px = sx[il], py = sy[il], pz = sz[il];
    float bd[KNN];
    int   bj[KNN];
#pragma unroll
    for (int m = 0; m < KNN; ++m) { bd[m] = 3.4e38f; bj[m] = il; }
    for (int j = 0; j < NPTS; ++j) {
        if (j == il) continue;
        float dx = sx[j] - px, dy = sy[j] - py, dz = sz[j] - pz;
        float d2 = dx * dx + dy * dy + dz * dz;
        if (d2 < bd[KNN - 1]) {
            int m = KNN - 1;
            while (m > 0 && bd[m - 1] > d2) { bd[m] = bd[m - 1]; bj[m] = bj[m - 1]; --m; }
            bd[m] = d2; bj[m] = j;
        }
    }
#pragma unroll
    for (int m = 0; m < KNN; ++m) {
        int j = bj[m];
        idx[i * KNN + m] = bb + j;   // global row index for direct gather
        rel[(size_t)(i * KNN + m) * 3 + 0] = (_Float16)(sx[j] - px);
        rel[(size_t)(i * KNN + m) * 3 + 1] = (_Float16)(sy[j] - py);
        rel[(size_t)(i * KNN + m) * 3 + 2] = (_Float16)(sz[j] - pz);
    }
}

// ---------------- LayerNorm (one wave per point, 4 dims per lane) ----------------
DEVFN float wave_sum(float s) {
#pragma unroll
    for (int o = 16; o > 0; o >>= 1) s += __shfl_xor(s, o, 32);
    return s;
}

template <bool OUT32>
__global__ __launch_bounds__(256)
void k_ln(const float* __restrict__ x, const float* __restrict__ g,
          const float* __restrict__ bta, _Float16* __restrict__ oh,
          float* __restrict__ of) {
    int lane = threadIdx.x & 31;
    int wave = threadIdx.x >> 5;
    int p    = blockIdx.x * 8 + wave;
    int d0   = lane * 4;
    f4 v  = *(const f4*)(x + (size_t)p * DIM + d0);
    float s = v[0] + v[1] + v[2] + v[3];
    float mu = wave_sum(s) * (1.f / DIM);
    float d[4], ss = 0.f;
#pragma unroll
    for (int j = 0; j < 4; ++j) { d[j] = v[j] - mu; ss += d[j] * d[j]; }
    float r = rsqrtf(wave_sum(ss) * (1.f / DIM) + LN_EPS);
    f4 gg = *(const f4*)(g + d0);
    f4 bb = *(const f4*)(bta + d0);
    if (OUT32) {
        f4 o;
#pragma unroll
        for (int j = 0; j < 4; ++j) o[j] = d[j] * r * gg[j] + bb[j];
        *(f4*)(of + (size_t)p * DIM + d0) = o;
    } else {
        h4 o;
#pragma unroll
        for (int j = 0; j < 4; ++j) o[j] = (_Float16)(d[j] * r * gg[j] + bb[j]);
        *(h4*)(oh + (size_t)p * DIM + d0) = o;
    }
}

// ---------------- stem (VALU; tiny fraction of FLOPs) ----------------
DEVFN void ln4(const float* v, const float* g, const float* bta, int d0, float* out) {
    float s = v[0] + v[1] + v[2] + v[3];
    float mu = wave_sum(s) * (1.f / DIM);
    float d[4], ss = 0.f;
#pragma unroll
    for (int j = 0; j < 4; ++j) { d[j] = v[j] - mu; ss += d[j] * d[j]; }
    float r = rsqrtf(wave_sum(ss) * (1.f / DIM) + LN_EPS);
#pragma unroll
    for (int j = 0; j < 4; ++j) {
        float y = d[j] * r * g[d0 + j] + bta[d0 + j];
        out[j] = y > 0.f ? y : 0.f;   // LN then ReLU
    }
}

__global__ __launch_bounds__(256)
void k_stem(const float* __restrict__ pts, const float* __restrict__ W1,
            const float* __restrict__ g1, const float* __restrict__ b1,
            const float* __restrict__ W2, const float* __restrict__ g2,
            const float* __restrict__ b2, float* __restrict__ feats) {
    __shared__ float sh[8][DIM];
    int lane = threadIdx.x & 31;
    int wave = threadIdx.x >> 5;
    int p    = blockIdx.x * 8 + wave;
    int d0   = lane * 4;
    float pc[6];
#pragma unroll
    for (int c = 0; c < 6; ++c) pc[c] = pts[(size_t)p * 6 + c];
    float h1[4];
#pragma unroll
    for (int j = 0; j < 4; ++j) {
        float s = 0.f;
#pragma unroll
        for (int c = 0; c < 6; ++c) s += pc[c] * W1[c * DIM + d0 + j];
        h1[j] = s;
    }
    float a1[4];
    ln4(h1, g1, b1, d0, a1);
#pragma unroll
    for (int j = 0; j < 4; ++j) sh[wave][d0 + j] = a1[j];
    __syncthreads();
    float h2[4] = {0.f, 0.f, 0.f, 0.f};
    for (int k = 0; k < DIM; ++k) {
        float hv = sh[wave][k];
#pragma unroll
        for (int j = 0; j < 4; ++j) h2[j] += hv * W2[k * DIM + d0 + j];
    }
    float a2[4];
    ln4(h2, g2, b2, d0, a2);
    f4 o; o[0] = a2[0]; o[1] = a2[1]; o[2] = a2[2]; o[3] = a2[3];
    *(f4*)(feats + (size_t)p * DIM + d0) = o;
}

// ---------------- generic WMMA GEMM: out = op(A[M x K] @ B[K x N]) ----------------
// One wave per 16-row M tile; A fragments hoisted in VGPRs; B tiles = 32B loads.
template <int KT, int NT, bool RELU, bool ACC, bool OUT16>
__global__ __launch_bounds__(128)
void k_gemm(const _Float16* __restrict__ A, const _Float16* __restrict__ Bw,
            float* __restrict__ Cf, _Float16* __restrict__ Ch) {
    const int Kd = KT * 32, Nd = NT * 16;
    int lane = threadIdx.x & 31;
    int wave = threadIdx.x >> 5;
    int row0 = (blockIdx.x * 4 + wave) * 16;

    h16 af[KT];
#pragma unroll
    for (int kt = 0; kt < KT; ++kt)
        af[kt] = load_a_frag(A + (size_t)row0 * Kd, Kd, lane, kt * 32);

#pragma unroll
    for (int nt = 0; nt < NT; ++nt) {
        f8 acc = {};
#pragma unroll
        for (int kt = 0; kt < KT; ++kt)
            acc = wmma16(af[kt], load_b_frag(Bw, Nd, lane, kt * 32, nt * 16), acc);
        int mrow = row0 + ((lane < 16) ? 0 : 8);
        int col  = nt * 16 + (lane & 15);
#pragma unroll
        for (int v = 0; v < 8; ++v) {
            float x = acc[v];
            if (RELU) x = x > 0.f ? x : 0.f;
            size_t off = (size_t)(mrow + v) * Nd + col;
            if (OUT16)      Ch[off]  = (_Float16)x;
            else if (ACC)   Cf[off] += x;
            else            Cf[off]  = x;
        }
    }
}

// ---------------- message MLP: one wave per point ----------------
// GEMM1 A = [neigh(128) | rel(3) | pad] 16x160, fragments gathered DIRECTLY from
// global (lane = neighbor row); rel/pad K-tile built in registers.
// Acc seeded with hw = h @ (W1a-W1c).  Only the GEMM1->GEMM2 transpose uses LDS.
__global__ __launch_bounds__(128)
void k_msg(const _Float16* __restrict__ h, const float* __restrict__ hw,
           const int* __restrict__ idx, const _Float16* __restrict__ rel,
           const _Float16* __restrict__ Wn, const _Float16* __restrict__ W2,
           _Float16* __restrict__ agg) {
    __shared__ _Float16 Ms[4][16 * DIM];
    int lane = threadIdx.x & 31;
    int wave = threadIdx.x >> 5;
    int i    = blockIdx.x * 4 + wave;       // point id
    _Float16* Mid = Ms[wave];

    // --- A fragments for GEMM1, hoisted across all N tiles ---
    // kt 0..3: gathered neighbor features, lane's row = h[idx[i][lane&15]]
    int nb = idx[i * KNN + (lane & 15)];
    const _Float16* rowp = h + (size_t)nb * DIM;
    h16 af1[5];
#pragma unroll
    for (int kt = 0; kt < 4; ++kt)
        af1[kt] = load_a_frag_row(rowp, lane, kt * 32);
    // kt 4: cols 128..159 = [rel_xyz(3) | zeros]; only lanes 0..15 (hi=0) carry data
    {
        h16 rf = {};
        if (lane < 16) {
            const _Float16* rp = rel + (size_t)(i * KNN + lane) * 3;
            rf[0] = rp[0]; rf[1] = rp[1]; rf[2] = rp[2];
        }
        af1[4] = rf;
    }

    // --- GEMM1: (16x160) @ Wn(160x128), C seeded with hw broadcast; ReLU -> Mid ---
#pragma unroll
    for (int nt = 0; nt < 8; ++nt) {
        float hv = hw[(size_t)i * DIM + nt * 16 + (lane & 15)];
        f8 acc;
#pragma unroll
        for (int v = 0; v < 8; ++v) acc[v] = hv;
#pragma unroll
        for (int kt = 0; kt < 5; ++kt)
            acc = wmma16(af1[kt], load_b_frag(Wn, DIM, lane, kt * 32, nt * 16), acc);
        int mrow = (lane < 16) ? 0 : 8;
        int col  = nt * 16 + (lane & 15);
#pragma unroll
        for (int v = 0; v < 8; ++v) {
            float x = acc[v];
            Mid[(mrow + v) * DIM + col] = (_Float16)(x > 0.f ? x : 0.f);
        }
    }
    // wave-private LDS region; DS ops are in-order within a wave.

    // --- GEMM2: (16x128) @ W2(128x128); relu + max over 16 neighbor rows -> agg ---
    h16 af2[4];
#pragma unroll
    for (int kt = 0; kt < 4; ++kt)
        af2[kt] = load_a_frag(Mid, DIM, lane, kt * 32);

#pragma unroll
    for (int nt = 0; nt < 8; ++nt) {
        f8 acc = {};
#pragma unroll
        for (int kt = 0; kt < 4; ++kt)
            acc = wmma16(af2[kt], load_b_frag(W2, DIM, lane, kt * 32, nt * 16), acc);
        float mx = 0.f;                        // relu folds into max-with-0
#pragma unroll
        for (int v = 0; v < 8; ++v) mx = fmaxf(mx, acc[v]);
        mx = fmaxf(mx, __shfl_xor(mx, 16, 32));   // combine M halves
        if (lane < 16)
            agg[(size_t)i * DIM + nt * 16 + lane] = (_Float16)mx;
    }
}

// ---------------- final pooling: mean & max over N per batch ----------------
__global__ __launch_bounds__(128)
void k_pool(const float* __restrict__ x, float* __restrict__ out) {
    int b = blockIdx.x, d = threadIdx.x;
    const float* xb = x + (size_t)b * NPTS * DIM;
    float s = 0.f, mx = -3.4e38f;
    for (int n = 0; n < NPTS; ++n) {
        float v = xb[(size_t)n * DIM + d];
        s += v; mx = fmaxf(mx, v);
    }
    out[b * 256 + d]       = s * (1.f / NPTS);
    out[b * 256 + 128 + d] = mx;
}

// ---------------- host ----------------
extern "C" void kernel_launch(void* const* d_in, const int* in_sizes, int n_in,
                              void* d_out, int out_size, void* d_ws, size_t ws_size,
                              hipStream_t stream) {
    const float* points  = (const float*)d_in[0];
    const float* stem_W1 = (const float*)d_in[1];
    const float* stem_g1 = (const float*)d_in[2];
    const float* stem_b1 = (const float*)d_in[3];
    const float* stem_W2 = (const float*)d_in[4];
    const float* stem_g2 = (const float*)d_in[5];
    const float* stem_b2 = (const float*)d_in[6];
    const float* ln1_g   = (const float*)d_in[7];
    const float* ln1_b   = (const float*)d_in[8];
    const float* msg_W1  = (const float*)d_in[9];
    const float* msg_W2  = (const float*)d_in[10];
    const float* proj_W  = (const float*)d_in[11];
    const float* ln2_g   = (const float*)d_in[12];
    const float* ln2_b   = (const float*)d_in[13];
    const float* ffn_W1  = (const float*)d_in[14];
    const float* ffn_W2  = (const float*)d_in[15];
    const float* norm_g  = (const float*)d_in[16];
    const float* norm_b  = (const float*)d_in[17];

    char* w = (char*)d_ws;
    size_t off = 0;
    auto alloc = [&](size_t bytes) -> void* {
        void* p = w + off;
        off = (off + bytes + 255) & ~(size_t)255;
        return p;
    };
    float*    feats = (float*)   alloc((size_t)PTOT * DIM * 4);
    _Float16* hbuf  = (_Float16*)alloc((size_t)PTOT * DIM * 2);
    float*    hw    = (float*)   alloc((size_t)PTOT * DIM * 4);
    _Float16* agg   = (_Float16*)alloc((size_t)PTOT * DIM * 2);
    _Float16* tbuf  = (_Float16*)alloc((size_t)PTOT * HIDDEN * 2);
    float*    xfin  = (float*)   alloc((size_t)PTOT * DIM * 4);
    int*      idxb  = (int*)     alloc((size_t)PTOT * KNN * 4);
    _Float16* relb  = (_Float16*)alloc((size_t)PTOT * KNN * 3 * 2);
    _Float16* Wh    = (_Float16*)alloc((size_t)DEPTH * 128 * 128 * 2);
    _Float16* Wn    = (_Float16*)alloc((size_t)DEPTH * 160 * 128 * 2);
    _Float16* W2c   = (_Float16*)alloc((size_t)DEPTH * 128 * 128 * 2);
    _Float16* Wpc   = (_Float16*)alloc((size_t)DEPTH * 128 * 128 * 2);
    _Float16* Wf1   = (_Float16*)alloc((size_t)DEPTH * 128 * 256 * 2);
    _Float16* Wf2   = (_Float16*)alloc((size_t)DEPTH * 256 * 128 * 2);
    (void)ws_size; (void)in_sizes; (void)n_in; (void)out_size;

    // ---- weight prep ----
    k_prep_msg1<<<(DEPTH * 160 * 128 + 255) / 256, 256, 0, stream>>>(msg_W1, Wh, Wn);
    k_cast16<<<(DEPTH * 128 * 128 + 255) / 256, 256, 0, stream>>>(msg_W2, W2c, DEPTH * 128 * 128);
    k_cast16<<<(DEPTH * 128 * 128 + 255) / 256, 256, 0, stream>>>(proj_W, Wpc, DEPTH * 128 * 128);
    k_cast16<<<(DEPTH * 128 * 256 + 255) / 256, 256, 0, stream>>>(ffn_W1, Wf1, DEPTH * 128 * 256);
    k_cast16<<<(DEPTH * 256 * 128 + 255) / 256, 256, 0, stream>>>(ffn_W2, Wf2, DEPTH * 256 * 128);

    // ---- geometry + stem ----
    k_knn<<<PTOT / 256, 256, 0, stream>>>(points, idxb, relb);
    k_stem<<<PTOT / 8, 256, 0, stream>>>(points, stem_W1, stem_g1, stem_b1,
                                         stem_W2, stem_g2, stem_b2, feats);

    // ---- transformer blocks ----
    for (int l = 0; l < DEPTH; ++l) {
        k_ln<false><<<PTOT / 8, 256, 0, stream>>>(feats, ln1_g + l * DIM, ln1_b + l * DIM,
                                                  hbuf, nullptr);
        // hw = h @ (W1a - W1c)
        k_gemm<4, 8, false, false, false><<<PTOT / 64, 128, 0, stream>>>(
            hbuf, Wh + (size_t)l * 128 * 128, hw, nullptr);
        // per-point message MLP + max-pool over neighbors
        k_msg<<<PTOT / 4, 128, 0, stream>>>(hbuf, hw, idxb, relb,
                                            Wn + (size_t)l * 160 * 128,
                                            W2c + (size_t)l * 128 * 128, agg);
        // feats += agg @ proj_W
        k_gemm<4, 8, false, true, false><<<PTOT / 64, 128, 0, stream>>>(
            agg, Wpc + (size_t)l * 128 * 128, feats, nullptr);
        // FFN
        k_ln<false><<<PTOT / 8, 256, 0, stream>>>(feats, ln2_g + l * DIM, ln2_b + l * DIM,
                                                  hbuf, nullptr);
        k_gemm<4, 16, true, false, true><<<PTOT / 64, 128, 0, stream>>>(
            hbuf, Wf1 + (size_t)l * 128 * 256, nullptr, tbuf);
        k_gemm<8, 8, false, true, false><<<PTOT / 64, 128, 0, stream>>>(
            tbuf, Wf2 + (size_t)l * 256 * 128, feats, nullptr);
    }

    // ---- final LN + pooling ----
    k_ln<true><<<PTOT / 8, 256, 0, stream>>>(feats, norm_g, norm_b, nullptr, xfin);
    k_pool<<<BATCH, 128, 0, stream>>>(xfin, (float*)d_out);
}
